// TitansMAC_29755533427430
// MI455X (gfx1250) — matrix operand order repaired and checked by reading, not compiled, not run on hardware
//
#include <hip/hip_runtime.h>
#include <math.h>
#include <stdint.h>

// ---------------- problem constants ----------------
#define B_    8
#define NTOK  2048
#define DD    512
#define DD4   2048
#define SEGL  128
#define PTL   16
#define LCTX  272          // PTL + 2*SEGL
#define NH    4
#define HD    128
#define NSEG  (NTOK / SEGL)
#define PADK  288          // 272 padded up to multiple of 32 (9 k-steps)
#define VTW   (B_ * LCTX + 16)   // 2192, padded width of transposed-V buffer

typedef __attribute__((ext_vector_type(16))) _Float16 v16h;
typedef __attribute__((ext_vector_type(8)))  _Float16 v8h;
typedef __attribute__((ext_vector_type(8)))  float    v8f;
typedef __attribute__((ext_vector_type(4)))  unsigned int v4u;
typedef __attribute__((ext_vector_type(8)))  int v8i;
typedef __attribute__((ext_vector_type(4)))  int v4i;

#if __has_builtin(__builtin_amdgcn_tensor_load_to_lds)
#define HAVE_TDM 1
#if __has_include(<hip/amd_detail/amd_gfx1250_TDM.h>)
#define TDM_ARGS6 1
#endif
#endif

// ---------------- epilogue flags ----------------
#define F_BIAS  1
#define F_SILU  2
#define F_C32   4
#define F_C16   8
#define F_CT16  16
#define F_DIAG  32
#define F_RESID 64

#define WMMA_F16(a, b, c) \
  __builtin_amdgcn_wmma_f32_16x16x32_f16(false, (a), false, (b), (short)0, (c), false, false)

// ---------------- WMMA fragment loads (gfx1250 wave32 layouts) ----------------
// A (16x32 f16, row-major source): lane l holds row M=l%16;
// lanes 0-15 take K {0..15}, lanes 16-31 take K {16..31}.
__device__ __forceinline__ v16h load_a_frag(const _Float16* A, int lda, int m0, int k0, int lane) {
  int r = lane & 15;
  const _Float16* p = A + (size_t)(m0 + r) * lda + k0 + ((lane >> 4) << 4);
  v8h lo = *(const v8h*)p;
  v8h hi = *(const v8h*)(p + 8);
  return __builtin_shufflevector(lo, hi, 0,1,2,3,4,5,6,7,8,9,10,11,12,13,14,15);
}
// B (32x16 f16) loaded from Bt = B^T (N x K row-major): lane l holds column N=l%16;
// lanes 0-15 hold K=k0..k0+15, lanes 16-31 hold K=k0+16..k0+31. One 32B load.
__device__ __forceinline__ v16h load_b_frag(const _Float16* Bt, int ldb, int n0, int k0, int lane) {
  const _Float16* p = Bt + (size_t)(n0 + (lane & 15)) * ldb + k0 + ((lane >> 4) << 4);
  return *(const v16h*)p;
}

// ---------------- TDM: async 2D panel load (64 rows x 32 f16 cols) into LDS ----------------
#if defined(HAVE_TDM)
__device__ __forceinline__ void tdm_load_panel(unsigned lds_off, const _Float16* gsrc, int ldb_elems) {
  unsigned long long ga = (unsigned long long)(uintptr_t)gsrc;
  v4u g0;
  g0[0] = 1u;                                              // count=1, user mode
  g0[1] = lds_off;                                         // LDS byte address
  g0[2] = (unsigned)(ga & 0xFFFFFFFFu);                    // global_addr[31:0]
  g0[3] = (unsigned)((ga >> 32) & 0x01FFFFFFu) | (2u << 30); // global_addr[56:32] | type=2
  v8i g1;
  g1[0] = (int)(1u << 16);                                 // workgroup_mask=0, data_size=2B
  g1[1] = (int)(((unsigned)ldb_elems & 0xFFFFu) << 16);    // tensor_dim0[15:0]
  g1[2] = (int)((((unsigned)ldb_elems >> 16) & 0xFFFFu) | (64u << 16)); // dim0[31:16] | tensor_dim1=64
  g1[3] = (int)(32u << 16);                                // tensor_dim1[31:16]=0 | tile_dim0=32
  g1[4] = 64;                                              // tile_dim1=64, tile_dim2=0
  g1[5] = ldb_elems;                                       // tensor_dim0_stride[31:0] (elements)
  g1[6] = 0;                                               // stride0[47:32]=0, dim1_stride[15:0]=0
  g1[7] = 0;
  v4i z4 = {0, 0, 0, 0};
#if defined(TDM_ARGS6)
  v8i z8 = {0, 0, 0, 0, 0, 0, 0, 0};
  __builtin_amdgcn_tensor_load_to_lds(g0, g1, z4, z4, z8, 0);
#else
  __builtin_amdgcn_tensor_load_to_lds(g0, g1, z4, z4, 0);
#endif
}
#endif

// ---------------- generic batched WMMA GEMM ----------------
// Macro-tile 64(M)x64(N) per workgroup of 4 waves; wave w computes rows [w*16, w*16+16).
// B panel (64 x 32 f16) staged in LDS (TDM double-buffered), shared by all 4 waves;
// A fragment reused across the 4 N-subtiles in registers.
struct GP {
  const _Float16* A;   long long sAy, sAz;  int lda;
  const _Float16* Bt;  long long sBy, sBz;  int ldb;
  const float*    bias;
  float*          C;   long long sCy, sCz;  int ldc;
  _Float16*       C16; long long sC16y, sC16z; int ldc16;
  _Float16*       Ct;  long long sCty, sCtz;   int ldct;
  const float*    R;   long long sRy, sRz;  int ldr;
  const float*    diag; int rpb;
  int M, N, K, flags;
};

__global__ __launch_bounds__(128) void gemm_wmma(GP p) {
  __shared__ __align__(64) _Float16 ldsB[2][64 * 32];
  const int wave = threadIdx.x >> 5;
  const int lane = threadIdx.x & 31;
  const int tnn = p.N >> 6;
  const int mt = (int)blockIdx.x / tnn, nt = (int)blockIdx.x % tnn;
  const int mBase = (mt << 6) + (wave << 4);
  const int n0 = nt << 6;

  const _Float16* A  = p.A  + blockIdx.y * p.sAy + blockIdx.z * p.sAz;
  const _Float16* Bt = p.Bt + blockIdx.y * p.sBy + blockIdx.z * p.sBz;
  const _Float16* Bpanel = Bt + (size_t)n0 * p.ldb;

  const int nk = p.K >> 5;

#if defined(HAVE_TDM)
  const unsigned ldsbase = (unsigned)(uintptr_t)(void*)&ldsB[0][0];
  if (wave == 0) tdm_load_panel(ldsbase, Bpanel, p.ldb);
#endif

  v8f acc[4] = {};
  for (int kt = 0; kt < nk; ++kt) {
    const int k0 = kt << 5;
    const int cur = kt & 1;
#if defined(HAVE_TDM)
    if (wave == 0) __builtin_amdgcn_s_wait_tensorcnt((short)0);
    __syncthreads();                 // publish buf[cur]; prior reads of buf[cur^1] done
    if (wave == 0 && kt + 1 < nk)
      tdm_load_panel(ldsbase + (unsigned)((cur ^ 1) * (64 * 32 * 2)),
                     Bpanel + k0 + 32, p.ldb);
#else
    __syncthreads();                 // prior reads of buf[cur] done
    {
      int idx = (int)threadIdx.x * 16;          // 128 thr x 16 halves = 64x32 panel
      int row = idx >> 5, col = idx & 31;
      *(v8h*)&ldsB[cur][idx]     = *(const v8h*)(Bpanel + (size_t)row * p.ldb + k0 + col);
      *(v8h*)&ldsB[cur][idx + 8] = *(const v8h*)(Bpanel + (size_t)row * p.ldb + k0 + col + 8);
    }
    __syncthreads();
#endif
    v16h a = load_a_frag(A, p.lda, mBase, k0, lane);
#pragma unroll
    for (int j = 0; j < 4; ++j) {
      const int r = (j << 4) + (lane & 15);
      const int c = (lane >> 4) << 4;
      v16h b = *(const v16h*)&ldsB[cur][r * 32 + c];
      acc[j] = WMMA_F16(a, b, acc[j]);
    }
  }

  // C/D layout: VGPR r -> M = m0 + r + (lane>=16 ? 8 : 0), N = n0 + lane%16
  const int rbase = mBase + ((lane >> 4) << 3);
  float*     C   = p.C   ? p.C   + blockIdx.y * p.sCy   + blockIdx.z * p.sCz   : nullptr;
  _Float16*  C16 = p.C16 ? p.C16 + blockIdx.y * p.sC16y + blockIdx.z * p.sC16z : nullptr;
  _Float16*  Ct  = p.Ct  ? p.Ct  + blockIdx.y * p.sCty  + blockIdx.z * p.sCtz  : nullptr;
  const float* R = p.R   ? p.R   + blockIdx.y * p.sRy   + blockIdx.z * p.sRz   : nullptr;

#pragma unroll
  for (int j = 0; j < 4; ++j) {
    const int cn = n0 + (j << 4) + (lane & 15);
#pragma unroll
    for (int r = 0; r < 8; ++r) {
      int gm = rbase + r;
      float v = acc[j][r];
      if (p.flags & F_BIAS)  v += p.bias[cn];
      if (p.flags & F_DIAG)  v *= p.diag[(gm / p.rpb) * p.N + cn];
      if (p.flags & F_SILU)  v = v / (1.0f + __expf(-v));
      if (p.flags & F_RESID) v += R[(size_t)gm * p.ldr + cn];
      if (p.flags & F_C32)   C[(size_t)gm * p.ldc + cn] = v;
      if (p.flags & F_C16)   C16[(size_t)gm * p.ldc16 + cn] = (_Float16)v;
      if (p.flags & F_CT16)  Ct[(size_t)cn * p.ldct + gm] = (_Float16)v;
    }
  }
}

// ---------------- attention scores (WMMA, causal mask + scale + pad) ----------------
__global__ __launch_bounds__(128) void attn_scores(const _Float16* q16, const _Float16* k16,
                                                   float* logits, float scale) {
  const int h = blockIdx.y, b = blockIdx.z;
  const int wave = threadIdx.x >> 5, lane = threadIdx.x & 31;
  const int TN = PADK / 16;                 // 18 key tiles
  const int tile = blockIdx.x * 4 + wave;   // 8*18 = 144 tiles
  if (tile >= 8 * TN) return;
  const int ti = tile / TN, tj = tile % TN;
  float* out = logits + (size_t)(b * NH + h) * SEGL * PADK;
  const int m0 = ti << 4, n0 = tj << 4;
  const int cn = n0 + (lane & 15);
  const int rbase = m0 + ((lane >> 4) << 3);
  if (tj == TN - 1) {                       // pure pad tile (keys 272..287)
#pragma unroll
    for (int r = 0; r < 8; ++r) out[(size_t)(rbase + r) * PADK + cn] = -1e30f;
    return;
  }
  const _Float16* qp = q16 + ((size_t)b * LCTX + PTL + SEGL) * DD + h * HD; // query rows 144..271
  const _Float16* kp = k16 + (size_t)b * LCTX * DD + h * HD;
  v8f acc = {};
  for (int k0 = 0; k0 < HD; k0 += 32) {
    v16h a  = load_a_frag(qp, DD, m0, k0, lane);
    v16h bb = load_b_frag(kp, DD, n0, k0, lane);
    acc = WMMA_F16(a, bb, acc);
  }
#pragma unroll
  for (int r = 0; r < 8; ++r) {
    int gm = rbase + r;                     // local query; ctx row = 144+gm
    float v = (cn <= PTL + SEGL + gm) ? acc[r] * scale : -1e30f;
    out[(size_t)gm * PADK + cn] = v;
  }
}

// ---------------- wave-per-row softmax over 288 cols ----------------
__global__ __launch_bounds__(32) void softmax_rows(const float* logits, _Float16* P) {
  const size_t row = blockIdx.x;
  const float* in = logits + row * PADK;
  _Float16* out = P + row * PADK;
  const int lane = threadIdx.x;
  float vals[9], mx = -1e30f;
#pragma unroll
  for (int i = 0; i < 9; ++i) { vals[i] = in[lane + i * 32]; mx = fmaxf(mx, vals[i]); }
  for (int m = 16; m; m >>= 1) mx = fmaxf(mx, __shfl_xor(mx, m, 32));
  float s = 0.f;
#pragma unroll
  for (int i = 0; i < 9; ++i) { vals[i] = __expf(vals[i] - mx); s += vals[i]; }
  for (int m = 16; m; m >>= 1) s += __shfl_xor(s, m, 32);
  const float inv = 1.0f / s;
#pragma unroll
  for (int i = 0; i < 9; ++i) out[lane + i * 32] = (_Float16)(vals[i] * inv);
}

// ---------------- wave-per-row layernorm (512 cols) ----------------
__global__ __launch_bounds__(32) void layernorm_rows(const float* in, float* out32, _Float16* out16,
                                                     const float* s, const float* bv) {
  const size_t row = blockIdx.x;
  const float* x = in + row * DD;
  const int lane = threadIdx.x;
  float vals[16], mu = 0.f;
#pragma unroll
  for (int i = 0; i < 16; ++i) { vals[i] = x[lane + i * 32]; mu += vals[i]; }
  for (int m = 16; m; m >>= 1) mu += __shfl_xor(mu, m, 32);
  mu *= (1.0f / DD);
  float var = 0.f;
#pragma unroll
  for (int i = 0; i < 16; ++i) { float d = vals[i] - mu; var += d * d; }
  for (int m = 16; m; m >>= 1) var += __shfl_xor(var, m, 32);
  var *= (1.0f / DD);
  const float inv = rsqrtf(var + 1e-6f);
#pragma unroll
  for (int i = 0; i < 16; ++i) {
    int c = lane + i * 32;
    float v = (vals[i] - mu) * inv * s[c] + bv[c];
    if (out32) out32[row * DD + c] = v;
    if (out16) out16[row * DD + c] = (_Float16)v;
  }
}

// ---------------- gates: sigmoid(y @ gate_w + gate_b), wave per token ----------------
__global__ __launch_bounds__(32) void gate_kernel(const float* y, const float* gw, const float* gb,
                                                  float* gates, long long strideB) {
  const int idx = blockIdx.x;               // b*SEGL + t
  const int b = idx / SEGL, t = idx % SEGL;
  const float* yr = y + (size_t)b * strideB + (size_t)t * DD;
  const int lane = threadIdx.x;
  float s0 = 0.f, s1 = 0.f, s2 = 0.f;
  for (int i = lane; i < DD; i += 32) {
    float v = yr[i];
    s0 += v * gw[i * 3 + 0]; s1 += v * gw[i * 3 + 1]; s2 += v * gw[i * 3 + 2];
  }
  for (int m = 16; m; m >>= 1) {
    s0 += __shfl_xor(s0, m, 32); s1 += __shfl_xor(s1, m, 32); s2 += __shfl_xor(s2, m, 32);
  }
  if (lane == 0) {
    gates[idx * 3 + 0] = 1.f / (1.f + __expf(-(s0 + gb[0])));
    gates[idx * 3 + 1] = 1.f / (1.f + __expf(-(s1 + gb[1])));
    gates[idx * 3 + 2] = 1.f / (1.f + __expf(-(s2 + gb[2])));
  }
}

// ---------------- diagonal fast-weight memory update (closed on diag(M),diag(S)) ----------------
__global__ __launch_bounds__(512) void mem_update(const float* kbuf, const float* vbuf,
                                                  const float* gates, float* diagM, float* diagS) {
  const int b = blockIdx.x, d = threadIdx.x;
  float mM = diagM[b * DD + d], mS = diagS[b * DD + d];
  for (int t = 0; t < SEGL; ++t) {
    int row = b * SEGL + t;
    float kt = kbuf[(size_t)row * DD + d];
    float vt = vbuf[(size_t)row * DD + d];
    float at = gates[row * 3 + 0], et = gates[row * 3 + 1], tt = gates[row * 3 + 2];
    float pred = kt * mM;
    float Sn = et * mS - tt * kt * (pred - vt);
    mM = (1.f - at) * mM + Sn;
    mS = Sn;
  }
  diagM[b * DD + d] = mM;
  diagS[b * DD + d] = mS;
}

// ---------------- small data-movement kernels ----------------
__global__ __launch_bounds__(256) void convert_seg(const float* x, _Float16* seg16, int s0) {
  size_t i = (size_t)blockIdx.x * 256 + threadIdx.x;
  if (i >= (size_t)B_ * SEGL * DD) return;
  int c = i % DD; size_t r = i / DD; int b = r / SEGL, t = r % SEGL;
  seg16[i] = (_Float16)x[((size_t)b * NTOK + s0 + t) * DD + c];
}

__global__ __launch_bounds__(256) void pack_ctx(const float* pmem, const _Float16* ht16,
                                                const _Float16* seg16, _Float16* ctx16) {
  size_t i = (size_t)blockIdx.x * 256 + threadIdx.x;
  if (i >= (size_t)B_ * LCTX * DD) return;
  int c = i % DD; size_t rr = i / DD; int b = rr / LCTX, r = rr % LCTX;
  _Float16 v;
  if (r < PTL)              v = (_Float16)pmem[r * DD + c];
  else if (r < PTL + SEGL)  v = ht16[((size_t)b * SEGL + (r - PTL)) * DD + c];
  else                      v = seg16[((size_t)b * SEGL + (r - PTL - SEGL)) * DD + c];
  ctx16[i] = v;
}

__global__ __launch_bounds__(256) void wtrans(const float* W, _Float16* Wt, int K, int N) {
  size_t i = (size_t)blockIdx.x * 256 + threadIdx.x;
  if (i >= (size_t)K * N) return;
  int n = i % N, k = i / N;
  Wt[(size_t)n * K + k] = (_Float16)W[i];
}

__global__ __launch_bounds__(256) void zero_f32(float* p, int n) {
  int i = blockIdx.x * 256 + threadIdx.x;
  if (i < n) p[i] = 0.f;
}

// ---------------- host ----------------
static void launch_gemm(hipStream_t st, const GP& p, int gy, int gz) {
  int tiles = (p.M >> 6) * (p.N >> 6);     // exact: M%64==0, N%64==0 for all call sites
  dim3 grid(tiles, gy, gz);
  gemm_wmma<<<grid, 128, 0, st>>>(p);
}

extern "C" void kernel_launch(void* const* d_in, const int* in_sizes, int n_in,
                              void* d_out, int out_size, void* d_ws, size_t ws_size,
                              hipStream_t stream) {
  (void)in_sizes; (void)n_in; (void)out_size; (void)ws_size;
  const float* x      = (const float*)d_in[0];
  const float* wk     = (const float*)d_in[1];
  const float* wv     = (const float*)d_in[2];
  const float* wq     = (const float*)d_in[3];
  const float* gate_w = (const float*)d_in[4];
  const float* gate_b = (const float*)d_in[5];
  const float* pmem   = (const float*)d_in[6];
  const float* aq_w   = (const float*)d_in[7];
  const float* aq_b   = (const float*)d_in[8];
  const float* ak_w   = (const float*)d_in[9];
  const float* ak_b   = (const float*)d_in[10];
  const float* av_w   = (const float*)d_in[11];
  const float* av_b   = (const float*)d_in[12];
  const float* ao_w   = (const float*)d_in[13];
  const float* ao_b   = (const float*)d_in[14];
  const float* ln1_s  = (const float*)d_in[15];
  const float* ln1_b  = (const float*)d_in[16];
  const float* ln2_s  = (const float*)d_in[17];
  const float* ln2_b  = (const float*)d_in[18];
  const float* f1_w   = (const float*)d_in[19];
  const float* f1_b   = (const float*)d_in[20];
  const float* f2_w   = (const float*)d_in[21];
  const float* f2_b   = (const float*)d_in[22];
  float* out = (float*)d_out;

  char* ws = (char*)d_ws;
  size_t off = 0;
  auto alloc = [&](size_t bytes) -> void* {
    void* p = ws + off;
    off = (off + bytes + 255) & ~(size_t)255;
    return p;
  };
  _Float16* wqT  = (_Float16*)alloc((size_t)DD * DD * 2);
  _Float16* aqT  = (_Float16*)alloc((size_t)DD * DD * 2);
  _Float16* akT  = (_Float16*)alloc((size_t)DD * DD * 2);
  _Float16* avT  = (_Float16*)alloc((size_t)DD * DD * 2);
  _Float16* aoT  = (_Float16*)alloc((size_t)DD * DD * 2);
  _Float16* wkT  = (_Float16*)alloc((size_t)DD * DD * 2);
  _Float16* wvT  = (_Float16*)alloc((size_t)DD * DD * 2);
  _Float16* f1T  = (_Float16*)alloc((size_t)DD4 * DD * 2);
  _Float16* f2T  = (_Float16*)alloc((size_t)DD * DD4 * 2);
  _Float16* seg16 = (_Float16*)alloc((size_t)B_ * SEGL * DD * 2);
  _Float16* ht16  = (_Float16*)alloc((size_t)B_ * SEGL * DD * 2);
  _Float16* ctx16 = (_Float16*)alloc((size_t)B_ * LCTX * DD * 2);
  _Float16* q16   = (_Float16*)alloc((size_t)B_ * LCTX * DD * 2);
  _Float16* k16   = (_Float16*)alloc((size_t)B_ * LCTX * DD * 2);
  _Float16* vT16  = (_Float16*)alloc((size_t)DD * VTW * 2);
  float*    logits = (float*)alloc((size_t)B_ * NH * SEGL * PADK * 4);
  _Float16* P16    = (_Float16*)alloc((size_t)B_ * NH * SEGL * PADK * 2);
  _Float16* attn16 = (_Float16*)alloc((size_t)B_ * SEGL * DD * 2);
  float*    y0     = (float*)alloc((size_t)B_ * SEGL * DD * 4);
  float*    y1     = (float*)alloc((size_t)B_ * SEGL * DD * 4);
  _Float16* y1n16  = (_Float16*)alloc((size_t)B_ * SEGL * DD * 2);
  _Float16* h1     = (_Float16*)alloc((size_t)B_ * SEGL * DD4 * 2);
  _Float16* y16    = (_Float16*)alloc((size_t)B_ * SEGL * DD * 2);
  float*    kbuf   = (float*)alloc((size_t)B_ * SEGL * DD * 4);
  float*    vbuf   = (float*)alloc((size_t)B_ * SEGL * DD * 4);
  float*    gates  = (float*)alloc((size_t)B_ * SEGL * 3 * 4);
  float*    diagM  = (float*)alloc((size_t)B_ * DD * 4);
  float*    diagS  = (float*)alloc((size_t)B_ * DD * 4);

  // weight prep (f16, transposed) + state init
  auto wt = [&](const float* W, _Float16* Wt, int K, int N) {
    wtrans<<<(int)(((size_t)K * N + 255) / 256), 256, 0, stream>>>(W, Wt, K, N);
  };
  wt(wq, wqT, DD, DD);   wt(aq_w, aqT, DD, DD); wt(ak_w, akT, DD, DD);
  wt(av_w, avT, DD, DD); wt(ao_w, aoT, DD, DD); wt(wk, wkT, DD, DD);
  wt(wv, wvT, DD, DD);   wt(f1_w, f1T, DD, DD4); wt(f2_w, f2T, DD4, DD);
  zero_f32<<<(B_ * DD + 255) / 256, 256, 0, stream>>>(diagM, B_ * DD);
  zero_f32<<<(B_ * DD + 255) / 256, 256, 0, stream>>>(diagS, B_ * DD);

  const float scale = 1.0f / sqrtf((float)HD);

  for (int s = 0; s < NSEG; ++s) {
    const int s0 = s * SEGL;

    convert_seg<<<(B_ * SEGL * DD) / 256, 256, 0, stream>>>(x, seg16, s0);

    // h_t = (seg @ wq) * diag(M)   -> ht16
    { GP p{}; p.A = seg16; p.lda = DD; p.Bt = wqT; p.ldb = DD;
      p.C16 = ht16; p.ldc16 = DD; p.diag = diagM; p.rpb = SEGL;
      p.M = B_ * SEGL; p.N = DD; p.K = DD; p.flags = F_DIAG | F_C16;
      launch_gemm(stream, p, 1, 1); }

    pack_ctx<<<(B_ * LCTX * DD) / 256, 256, 0, stream>>>(pmem, ht16, seg16, ctx16);

    // q/k projections (f16 out), v projection (transposed f16 out)
    { GP p{}; p.A = ctx16; p.lda = DD; p.Bt = aqT; p.ldb = DD; p.bias = aq_b;
      p.C16 = q16; p.ldc16 = DD; p.M = B_ * LCTX; p.N = DD; p.K = DD;
      p.flags = F_BIAS | F_C16; launch_gemm(stream, p, 1, 1); }
    { GP p{}; p.A = ctx16; p.lda = DD; p.Bt = akT; p.ldb = DD; p.bias = ak_b;
      p.C16 = k16; p.ldc16 = DD; p.M = B_ * LCTX; p.N = DD; p.K = DD;
      p.flags = F_BIAS | F_C16; launch_gemm(stream, p, 1, 1); }
    { GP p{}; p.A = ctx16; p.lda = DD; p.Bt = avT; p.ldb = DD; p.bias = av_b;
      p.Ct = vT16; p.ldct = VTW; p.M = B_ * LCTX; p.N = DD; p.K = DD;
      p.flags = F_BIAS | F_CT16; launch_gemm(stream, p, 1, 1); }

    attn_scores<<<dim3((8 * (PADK / 16) + 3) / 4, NH, B_), 128, 0, stream>>>(q16, k16, logits, scale);
    softmax_rows<<<B_ * NH * SEGL, 32, 0, stream>>>(logits, P16);

    // O = P @ V  (batched over h=grid.y, b=grid.z)
    { GP p{}; p.A = P16; p.lda = PADK; p.sAy = (long long)SEGL * PADK; p.sAz = (long long)NH * SEGL * PADK;
      p.Bt = vT16; p.ldb = VTW; p.sBy = (long long)HD * VTW; p.sBz = LCTX;
      p.C16 = attn16; p.ldc16 = DD; p.sC16y = HD; p.sC16z = (long long)SEGL * DD;
      p.M = SEGL; p.N = HD; p.K = PADK; p.flags = F_C16;
      launch_gemm(stream, p, NH, B_); }

    // ao projection + residual(seg)  -> y0 f32
    { GP p{}; p.A = attn16; p.lda = DD; p.sAy = (long long)SEGL * DD;
      p.Bt = aoT; p.ldb = DD; p.bias = ao_b;
      p.R = x + (size_t)s0 * DD; p.ldr = DD; p.sRy = (long long)NTOK * DD;
      p.C = y0; p.ldc = DD; p.sCy = (long long)SEGL * DD;
      p.M = SEGL; p.N = DD; p.K = DD; p.flags = F_BIAS | F_RESID | F_C32;
      launch_gemm(stream, p, B_, 1); }

    layernorm_rows<<<B_ * SEGL, 32, 0, stream>>>(y0, y1, nullptr, ln1_s, ln1_b);
    layernorm_rows<<<B_ * SEGL, 32, 0, stream>>>(y1, nullptr, y1n16, ln2_s, ln2_b);

    // FFN: h1 = silu(y1n @ f1 + b1); y = h1 @ f2 + b2 + y1 -> d_out (+f16 copy)
    { GP p{}; p.A = y1n16; p.lda = DD; p.Bt = f1T; p.ldb = DD; p.bias = f1_b;
      p.C16 = h1; p.ldc16 = DD4; p.M = B_ * SEGL; p.N = DD4; p.K = DD;
      p.flags = F_BIAS | F_SILU | F_C16; launch_gemm(stream, p, 1, 1); }
    { GP p{}; p.A = h1; p.lda = DD4; p.sAy = (long long)SEGL * DD4;
      p.Bt = f2T; p.ldb = DD4; p.bias = f2_b;
      p.R = y1; p.ldr = DD; p.sRy = (long long)SEGL * DD;
      p.C = out + (size_t)s0 * DD; p.ldc = DD; p.sCy = (long long)NTOK * DD;
      p.C16 = y16; p.ldc16 = DD; p.sC16y = (long long)SEGL * DD;
      p.M = SEGL; p.N = DD; p.K = DD4; p.flags = F_BIAS | F_RESID | F_C32 | F_C16;
      launch_gemm(stream, p, B_, 1); }

    gate_kernel<<<B_ * SEGL, 32, 0, stream>>>(out + (size_t)s0 * DD, gate_w, gate_b,
                                              gates, (long long)NTOK * DD);

    // k_ = y @ wk ; v_ = y @ wv
    { GP p{}; p.A = y16; p.lda = DD; p.Bt = wkT; p.ldb = DD;
      p.C = kbuf; p.ldc = DD; p.M = B_ * SEGL; p.N = DD; p.K = DD;
      p.flags = F_C32; launch_gemm(stream, p, 1, 1); }
    { GP p{}; p.A = y16; p.lda = DD; p.Bt = wvT; p.ldb = DD;
      p.C = vbuf; p.ldc = DD; p.M = B_ * SEGL; p.N = DD; p.K = DD;
      p.flags = F_C32; launch_gemm(stream, p, 1, 1); }

    mem_update<<<B_, 512, 0, stream>>>(kbuf, vbuf, gates, diagM, diagS);
  }
}
